// ViTVQ3D_15238543966223
// MI455X (gfx1250) — matrix operand order repaired and checked by reading, not compiled, not run on hardware
//
#include <hip/hip_runtime.h>
#include <hip/hip_bf16.h>
#include <stddef.h>

// ============================================================================
// ViT-VQ 3D autoencoder for MI455X (gfx1250, wave32, WMMA + TDM).
// Convs: implicit GEMM on v_wmma_f32_16x16x32_f16, block tile 32x256 (8 waves,
// each wave 16x64 = 4 accumulators). A (weight) tile DMA'd to LDS by the
// Tensor Data Mover (wave 0 issues, TENSORcnt wait), overlapped with the
// branch-free im2col B-tile build. LDS: A as f32 with TDM-generated 33-DWORD
// pitch (conflict-free), B as packed f16 pairs, column-major.
// ============================================================================

typedef __attribute__((ext_vector_type(16))) _Float16 v16h;
typedef __attribute__((ext_vector_type(8)))  float    v8f;
typedef __attribute__((ext_vector_type(4)))  unsigned uint32x4;
typedef __attribute__((ext_vector_type(8)))  int      int32x8;
typedef __attribute__((ext_vector_type(4)))  int      int32x4;

union FragU { v16h h; unsigned u[8]; };

static __device__ __forceinline__ unsigned pack2(float a, float b)
{
    _Float16 ha = (_Float16)a, hb = (_Float16)b;
    unsigned short ua = __builtin_bit_cast(unsigned short, ha);
    unsigned short ub = __builtin_bit_cast(unsigned short, hb);
    return (unsigned)ua | ((unsigned)ub << 16);
}

#if defined(__has_builtin)
#  if __has_builtin(__builtin_amdgcn_tensor_load_to_lds) && \
      __has_builtin(__builtin_amdgcn_s_wait_tensorcnt)
#    define USE_TDM 1
#  endif
#endif
#ifndef USE_TDM
#  define USE_TDM 0
#endif

#if USE_TDM
// DMA a 32x32-f32 2D tile (row pitch row_stride elems) into LDS at lds_addr,
// with 1-DWORD padding after every 32 DWORDs -> LDS pitch 33 f32.
// OOB rows/cols (beyond rows/cols_valid) are zero-filled by the TDM.
static __device__ __forceinline__ void tdm_load_a32x32(
    const float* gsrc, unsigned lds_addr, int rows, int cols_valid,
    long long row_stride_elems)
{
    unsigned long long ga = (unsigned long long)(size_t)gsrc;
    uint32x4 g0;
    g0[0] = 1u;                                           // count=1 (valid user D#)
    g0[1] = lds_addr;                                     // lds_addr [63:32]
    g0[2] = (unsigned)(ga & 0xFFFFFFFFu);                 // global_addr lo
    g0[3] = (unsigned)((ga >> 32) & 0x1FFFFFFu) | (2u << 30); // ga hi | type=2
    unsigned t0 = (unsigned)cols_valid;                   // tensor_dim0
    unsigned t1 = (unsigned)rows;                         // tensor_dim1
    unsigned long long s0 = (unsigned long long)row_stride_elems;
    int32x8 g1;
    g1[0] = (int)((2u << 16) | (1u << 20) | (4u << 22));  // 4B elems, pad_en, 32-DW interval (+1 DW)
    g1[1] = (int)((t0 & 0xFFFFu) << 16);                  // tensor_dim0 lo16
    g1[2] = (int)(((t0 >> 16) & 0xFFFFu) | ((t1 & 0xFFFFu) << 16));
    g1[3] = (int)(((t1 >> 16) & 0xFFFFu) | (32u << 16));  // tile_dim0 = 32
    g1[4] = (int)32u;                                     // tile_dim1 = 32, tile_dim2 = 0
    g1[5] = (int)(s0 & 0xFFFFFFFFu);                      // dim0_stride lo32
    g1[6] = (int)((s0 >> 32) & 0xFFFFu);                  // dim0_stride hi16
    g1[7] = 0;
    int32x4 z = {0, 0, 0, 0};
#if __clang_major__ >= 23
    int32x8 z8 = {0, 0, 0, 0, 0, 0, 0, 0};
    __builtin_amdgcn_tensor_load_to_lds(g0, g1, z, z, z8, 0);
#else
    __builtin_amdgcn_tensor_load_to_lds(g0, g1, z, z, 0);
#endif
}
#endif

#define CONV_THREADS 256   // 8 waves; wave grid 2(M) x 4(N)

// ========================= implicit-GEMM conv3d =============================
// normal conv (pad 1):  id = od*stride + kd - 1
// transposed (lhs_dilation=2, pad(1,2) == ConvTranspose3d k3 s2 p1 op1):
//   t = od + kd - 1; valid iff t>=0 && even && t/2 < Di ; id = t/2
__global__ __launch_bounds__(CONV_THREADS)
void conv3d_wmma(const float* __restrict__ x, const float* __restrict__ w,
                 const float* __restrict__ bias, float* __restrict__ y,
                 int Cin, int Cout,
                 int Di, int Hi, int Wi,
                 int Do, int Ho, int Wo,
                 int stride, int transposed)
{
    __shared__ float    Asf[32][33];     // f32 A tile, TDM-padded pitch 33
    __shared__ unsigned Bs32[256][17];   // packed f16 pairs, column-major

    const int tid   = threadIdx.x;
    const int wave  = tid >> 5;
    const int lane  = tid & 31;
    const int l     = lane & 15;
    const int hi    = lane >> 4;
    const int waveM = wave >> 2;
    const int waveN = wave & 3;
    const int tileM = blockIdx.x * 32;
    const int tileN = blockIdx.y * 256;
    const int Nsp   = Do * Ho * Wo;
    const int K     = Cin * 27;
    const int ksteps = (K + 31) >> 5;
    const int DHWi  = Di * Hi * Wi;

#if !USE_TDM
    const int arow = tid >> 3;
    const int aq   = tid & 7;
    const float* wrow =
        w + (size_t)((tileM + arow) < Cout ? (tileM + arow) : 0) * K;
#endif

    // ---- per-thread B-loader: one output column per thread ----
    const int nn = tileN + tid;
    const int od = nn / (Ho * Wo);
    const int oh = (nn / Wo) % Ho;
    const int ow = nn % Wo;
    const int odS = (transposed ? od : od * stride) - 1;   // + kd = src coord
    const int ohS = (transposed ? oh : oh * stride) - 1;
    const int owS = (transposed ? ow : ow * stride) - 1;
    int kcur = 0, kd = 0, kh = 0, kw = 0, ciOff = 0;       // uniform (SGPR)

    auto fetchB = [&]() -> float {
        bool ok = (kcur < K);
        int id, ih, iw;
        if (!transposed) {
            id = odS + kd; ih = ohS + kh; iw = owS + kw;
        } else {
            int td = odS + kd, th = ohS + kh, tw2 = owS + kw;
            ok = ok && !((td | th | tw2) & 1);
            id = td >> 1; ih = th >> 1; iw = tw2 >> 1;
        }
        ok = ok && (unsigned)id < (unsigned)Di
                && (unsigned)ih < (unsigned)Hi
                && (unsigned)iw < (unsigned)Wi;
        int idx = ciOff + (id * Hi + ih) * Wi + iw;
        float v = x[ok ? idx : 0];                         // load always in-bounds
        return ok ? v : 0.f;                               // v_cndmask, no branch
    };
    auto advB = [&]() {
        ++kcur; ++kw;
        bool cw = (kw == 3); kw = cw ? 0 : kw; kh += cw ? 1 : 0;
        bool ch = (kh == 3); kh = ch ? 0 : kh; kd += ch ? 1 : 0;
        bool cd = (kd == 3); kd = cd ? 0 : kd; ciOff += cd ? DHWi : 0;
    };

    v8f acc[4] = {};
    for (int ks = 0; ks < ksteps; ++ks) {
        const int k0 = ks << 5;
        // ---- A tile ----
#if USE_TDM
        if (wave == 0)
            tdm_load_a32x32(w + (size_t)tileM * K + k0,
                            (unsigned)(size_t)&Asf[0][0],
                            Cout - tileM, K - k0, K);
#else
        {
            int kb = k0 + aq * 4;
#pragma unroll
            for (int j = 0; j < 4; ++j) {
                int kc = (kb + j < K) ? (kb + j) : 0;
                float v = wrow[kc];
                Asf[arow][aq * 4 + j] = (kb + j < K) ? v : 0.f;
            }
        }
#endif
        // ---- B tile: 16 packed k-pairs per column (overlaps the TDM) ----
        for (int p = 0; p < 16; ++p) {
            float v0 = fetchB(); advB();
            float v1 = fetchB(); advB();
            Bs32[tid][p] = pack2(v0, v1);
        }
#if USE_TDM
        if (wave == 0) __builtin_amdgcn_s_wait_tensorcnt(0);
#endif
        __syncthreads();
        // ---- fragments + 4 WMMAs (A reused) ----
        FragU a;
        const float* arp = &Asf[waveM * 16 + l][0];
#pragma unroll
        for (int vi = 0; vi < 8; ++vi) {
            int ab = hi * 8 + vi;   // f16-pair index
            a.u[vi] = pack2(arp[2 * ab], arp[2 * ab + 1]);
        }
#pragma unroll
        for (int sub = 0; sub < 4; ++sub) {
            FragU b;
            int col = waveN * 64 + sub * 16 + l;
#pragma unroll
            for (int vi = 0; vi < 8; ++vi)
                b.u[vi] = Bs32[col][hi * 8 + vi];
            acc[sub] = __builtin_amdgcn_wmma_f32_16x16x32_f16(
                false, a.h, false, b.h, (short)0, acc[sub], false, false);
        }
        __syncthreads();
    }
#pragma unroll
    for (int sub = 0; sub < 4; ++sub) {
#pragma unroll
        for (int r = 0; r < 8; ++r) {
            int m = tileM + waveM * 16 + r + hi * 8;
            int n = tileN + waveN * 64 + sub * 16 + l;
            if (m < Cout && n < Nsp)
                y[(size_t)m * Nsp + n] = acc[sub][r] + bias[m];
        }
    }
}

// ===================== generic strided WMMA GEMM ============================
// C[m,n] = sum_k A[m*Asm + k*Ask] * B[k*Bsk + n*Bsn] (+ bias[n])
__global__ __launch_bounds__(CONV_THREADS)
void gemm_wmma(const float* __restrict__ A, const float* __restrict__ B,
               const float* __restrict__ bias, float* __restrict__ C,
               int M, int N, int K,
               long long Asm_, long long Ask,
               long long Bsk, long long Bsn,
               long long Csm, long long Csn)
{
    __shared__ unsigned As32[32][17];
    __shared__ unsigned Bs32[256][17];

    const int tid   = threadIdx.x;
    const int wave  = tid >> 5;
    const int lane  = tid & 31;
    const int l     = lane & 15;
    const int hi    = lane >> 4;
    const int waveM = wave >> 2;
    const int waveN = wave & 3;
    const int tileM = blockIdx.x * 32;
    const int tileN = blockIdx.y * 256;
    const int ksteps = (K + 31) >> 5;

    const int arow = tid >> 3;
    const int aq   = tid & 7;
    // Clamp OOB rows/cols to 0: their garbage only reaches masked D rows/cols.
    const float* Arow = A + (long long)((tileM + arow) < M ? (tileM + arow) : 0) * Asm_;
    const int bn = tileN + tid;
    const float* Bcol = B + (long long)(bn < N ? bn : 0) * Bsn;

    v8f acc[4] = {};
    for (int ks = 0; ks < ksteps; ++ks) {
        const int k0 = ks << 5;
        {
            int kb = k0 + aq * 4;
            float av[4];
#pragma unroll
            for (int j = 0; j < 4; ++j) {
                int kc = (kb + j < K) ? (kb + j) : 0;
                float v = Arow[(long long)kc * Ask];
                av[j] = (kb + j < K) ? v : 0.f;
            }
            As32[arow][aq * 2]     = pack2(av[0], av[1]);
            As32[arow][aq * 2 + 1] = pack2(av[2], av[3]);
        }
#pragma unroll
        for (int p = 0; p < 16; ++p) {
            int k = k0 + 2 * p;
            int kc0 = (k < K) ? k : 0;
            int kc1 = (k + 1 < K) ? (k + 1) : 0;
            float v0 = Bcol[(long long)kc0 * Bsk];
            float v1 = Bcol[(long long)kc1 * Bsk];
            v0 = (k < K) ? v0 : 0.f;
            v1 = (k + 1 < K) ? v1 : 0.f;
            Bs32[tid][p] = pack2(v0, v1);
        }
        __syncthreads();
        FragU a;
#pragma unroll
        for (int vi = 0; vi < 8; ++vi) {
            int ab = hi * 8 + vi;
            a.u[vi] = As32[waveM * 16 + l][ab];
        }
#pragma unroll
        for (int sub = 0; sub < 4; ++sub) {
            FragU b;
            int col = waveN * 64 + sub * 16 + l;
#pragma unroll
            for (int vi = 0; vi < 8; ++vi)
                b.u[vi] = Bs32[col][hi * 8 + vi];
            acc[sub] = __builtin_amdgcn_wmma_f32_16x16x32_f16(
                false, a.h, false, b.h, (short)0, acc[sub], false, false);
        }
        __syncthreads();
    }
#pragma unroll
    for (int sub = 0; sub < 4; ++sub) {
#pragma unroll
        for (int r = 0; r < 8; ++r) {
            int m = tileM + waveM * 16 + r + hi * 8;
            int n = tileN + waveN * 64 + sub * 16 + l;
            if (m < M && n < N) {
                float v = acc[sub][r];
                if (bias) v += bias[n];
                C[(long long)m * Csm + (long long)n * Csn] = v;
            }
        }
    }
}

// ================= InstanceNorm3d + PReLU (+ optional residual add) =========
__global__ void inorm_prelu(const float* __restrict__ x, const float* __restrict__ add,
                            float* __restrict__ y, int S, const float* __restrict__ alpha)
{
    const int c = blockIdx.x;
    const float* xc = x + (size_t)c * S;
    float* yc = y + (size_t)c * S;
    __shared__ float r1[256], r2[256];
    __shared__ float mv[2];
    float s = 0.f, s2 = 0.f;
    for (int i = threadIdx.x; i < S; i += 256) {
        float v = xc[i]; s += v; s2 += v * v;
    }
    r1[threadIdx.x] = s; r2[threadIdx.x] = s2;
    __syncthreads();
    for (int st = 128; st > 0; st >>= 1) {
        if (threadIdx.x < st) {
            r1[threadIdx.x] += r1[threadIdx.x + st];
            r2[threadIdx.x] += r2[threadIdx.x + st];
        }
        __syncthreads();
    }
    if (threadIdx.x == 0) {
        float m = r1[0] / S;
        float var = r2[0] / S - m * m;
        mv[0] = m; mv[1] = rsqrtf(var + 1e-5f);
    }
    __syncthreads();
    const float m = mv[0], rs = mv[1], a = *alpha;
    const float* ac = add ? add + (size_t)c * S : nullptr;
    for (int i = threadIdx.x; i < S; i += 256) {
        float v = (xc[i] - m) * rs;
        v = (v >= 0.f) ? v : a * v;
        if (ac) v += ac[i];
        yc[i] = v;
    }
}

// ====================== row-wise L2 normalization ===========================
__global__ void l2norm_rows(const float* __restrict__ x, float* __restrict__ y, int D)
{
    const int r = blockIdx.x;
    __shared__ float red[256];
    __shared__ float scale;
    float s = 0.f;
    for (int d = threadIdx.x; d < D; d += 256) {
        float v = x[(size_t)r * D + d]; s += v * v;
    }
    red[threadIdx.x] = s;
    __syncthreads();
    for (int st = 128; st > 0; st >>= 1) {
        if (threadIdx.x < st) red[threadIdx.x] += red[threadIdx.x + st];
        __syncthreads();
    }
    if (threadIdx.x == 0) scale = 1.f / fmaxf(sqrtf(red[0]), 1e-12f);
    __syncthreads();
    for (int d = threadIdx.x; d < D; d += 256)
        y[(size_t)r * D + d] = x[(size_t)r * D + d] * scale;
}

// =========== VQ: argmax over codes, gather q, commit partial sum ============
__global__ void vq_select(const float* __restrict__ sim, const float* __restrict__ en,
                          const float* __restrict__ h, float* __restrict__ q,
                          float* __restrict__ idx_out, float* __restrict__ accum,
                          int NCODE, int D)
{
    const int t = blockIdx.x;
    __shared__ float bv[256];
    __shared__ int   bi[256];
    float best = -3.4e38f; int besti = 0;
    for (int c = threadIdx.x; c < NCODE; c += 256) {
        float v = sim[(size_t)t * NCODE + c];
        if (v > best) { best = v; besti = c; }
    }
    bv[threadIdx.x] = best; bi[threadIdx.x] = besti;
    __syncthreads();
    for (int st = 128; st > 0; st >>= 1) {
        if (threadIdx.x < st) {
            float v2 = bv[threadIdx.x + st]; int i2 = bi[threadIdx.x + st];
            if (v2 > bv[threadIdx.x] ||
                (v2 == bv[threadIdx.x] && i2 < bi[threadIdx.x])) {
                bv[threadIdx.x] = v2; bi[threadIdx.x] = i2;
            }
        }
        __syncthreads();
    }
    const int sel = bi[0];
    if (threadIdx.x == 0) idx_out[t] = (float)sel;
    __shared__ float red[256];
    float part = 0.f;
    for (int d = threadIdx.x; d < D; d += 256) {
        float qv = en[(size_t)sel * D + d];
        q[(size_t)t * D + d] = qv;
        float df = qv - h[(size_t)t * D + d];
        part += df * df;
    }
    red[threadIdx.x] = part;
    __syncthreads();
    for (int st = 128; st > 0; st >>= 1) {
        if (threadIdx.x < st) red[threadIdx.x] += red[threadIdx.x + st];
        __syncthreads();
    }
    if (threadIdx.x == 0) atomicAdd(accum, red[0]);
}

__global__ void zero_scalar(float* p) { *p = 0.f; }
__global__ void commit_finalize(const float* accum, float* out, float inv_n)
{
    *out = (*accum) * inv_n;
}

// ============================================================================
// Host orchestration.
//
// Assumed d_in flattening (setup_inputs dict insertion order; params nested
// dicts/lists in insertion order):
//   [0] x (1,1,64,64,64)
//   enc stage s (s=0..2), base E = 1 + 15*s:
//     E+0..5 ws[0..5]  E+6..11 bs[0..5]  E+12 a(6)  E+13 res_w  E+14 res_b
//   dec stage s (s=0..2), base Dc = 46 + 6*s:
//     Dc+0 tw  Dc+1 tb  Dc+2 ta  Dc+3 rw  Dc+4 rb  Dc+5 ra
//   [64] pre_w  [65] pre_b  [66] post_w  [67] post_b  [68] codebook
//
// d_out (f32): [0..262143] g ; [262144..262655] idx (as float) ; [262656] commit
// Workspace requirement: ~110 MB.
// ============================================================================
extern "C" void kernel_launch(void* const* d_in, const int* in_sizes, int n_in,
                              void* d_out, int out_size, void* d_ws, size_t ws_size,
                              hipStream_t stream)
{
    (void)in_sizes; (void)n_in; (void)out_size; (void)ws_size;

    const float* x = (const float*)d_in[0];
    auto IN = [&](int i) { return (const float*)d_in[i]; };

    float* W     = (float*)d_ws;
    const size_t BIG = 8388608;          // 256 * 32^3
    float* bufA  = W;
    float* bufB  = bufA + BIG;
    float* bufC  = bufB + BIG;
    float* tok   = bufC + BIG;           // 512 x 1024  (h)
    float* hn    = tok  + 524288;
    float* enn   = hn   + 524288;
    float* simm  = enn  + 524288;        // 512 x 512
    float* qbuf  = simm + 262144;        // 512 x 1024
    float* accum = qbuf + 524288;        // 1

    float* outf       = (float*)d_out;
    float* g_out      = outf;            // 262144
    float* idx_out    = outf + 262144;   // 512
    float* commit_out = outf + 262656;   // 1

    auto conv = [&](const float* xin, const float* wp, const float* bp, float* yo,
                    int Cin, int Cout, int Din, int Dout, int stride, int transposed) {
        dim3 grid((Cout + 31) / 32, (Dout * Dout * Dout + 255) / 256);
        conv3d_wmma<<<grid, CONV_THREADS, 0, stream>>>(
            xin, wp, bp, yo, Cin, Cout, Din, Din, Din, Dout, Dout, Dout,
            stride, transposed);
    };
    auto inp = [&](const float* xin, const float* add, float* yo,
                   int C, int S, const float* alpha) {
        inorm_prelu<<<dim3(C), 256, 0, stream>>>(xin, add, yo, S, alpha);
    };

    // ================= encoder =================
    static const int eCin[3]  = {1, 256, 512};
    static const int eCout[3] = {256, 512, 1024};
    static const int eDin[3]  = {64, 32, 16};
    static const int eDout[3] = {32, 16, 8};

    const float* cur = x;
    for (int s = 0; s < 3; ++s) {
        const int E = 1 + 15 * s;
        const int Cin = eCin[s], Cout = eCout[s], Din = eDin[s], Dout = eDout[s];
        const int S = Dout * Dout * Dout;
        const float* a_p = IN(E + 12);
        conv(cur, IN(E + 13), IN(E + 14), bufC, Cin, Cout, Din, Dout, 2, 0);  // residual
        conv(cur, IN(E + 0), IN(E + 6), bufA, Cin, Cout, Din, Dout, 2, 0);    // subunit 0
        inp(bufA, nullptr, bufB, Cout, S, a_p + 0);
        for (int i = 1; i < 6; ++i) {
            conv(bufB, IN(E + i), IN(E + 6 + i), bufA, Cout, Cout, Dout, Dout, 1, 0);
            inp(bufA, (i == 5) ? bufC : nullptr, bufB, Cout, S, a_p + i);
        }
        cur = bufB;
    }

    // ================= VQ bottleneck =================
    gemm_wmma<<<dim3(512 / 32, 1024 / 256), CONV_THREADS, 0, stream>>>(
        bufB, IN(64), IN(65), tok, 512, 1024, 1024,
        1LL, 512LL, 1LL, 1024LL, 1024LL, 1LL);

    l2norm_rows<<<dim3(512), 256, 0, stream>>>(tok, hn, 1024);
    l2norm_rows<<<dim3(512), 256, 0, stream>>>(IN(68), enn, 1024);

    gemm_wmma<<<dim3(512 / 32, 512 / 256), CONV_THREADS, 0, stream>>>(
        hn, enn, nullptr, simm, 512, 512, 1024,
        1024LL, 1LL, 1LL, 1024LL, 512LL, 1LL);

    zero_scalar<<<1, 1, 0, stream>>>(accum);
    vq_select<<<dim3(512), 256, 0, stream>>>(simm, enn, tok, qbuf, idx_out, accum,
                                             512, 1024);
    commit_finalize<<<1, 1, 0, stream>>>(accum, commit_out,
                                         1.0f / (512.0f * 1024.0f));

    gemm_wmma<<<dim3(512 / 32, 1024 / 256), CONV_THREADS, 0, stream>>>(
        qbuf, IN(66), IN(67), bufA, 512, 1024, 1024,
        1024LL, 1LL, 1LL, 1024LL, 1LL, 512LL);

    // ================= decoder =================
    static const int dCin[3]  = {1024, 512, 256};
    static const int dCout[3] = {512, 256, 1};
    static const int dDin[3]  = {8, 16, 32};
    static const int dDout[3] = {16, 32, 64};

    for (int s = 0; s < 3; ++s) {
        const int Dc = 46 + 6 * s;
        const int Cin = dCin[s], Cout = dCout[s], Din = dDin[s], Dout = dDout[s];
        const int S = Dout * Dout * Dout;
        conv(bufA, IN(Dc + 0), IN(Dc + 1), bufB, Cin, Cout, Din, Dout, 2, 1);
        inp(bufB, nullptr, bufC, Cout, S, IN(Dc + 2));
        conv(bufC, IN(Dc + 3), IN(Dc + 4), bufB, Cout, Cout, Dout, Dout, 1, 0);
        float* stage_out = (s == 2) ? g_out : bufA;
        inp(bufB, bufC, stage_out, Cout, S, IN(Dc + 5));
    }
}
